// NewsEncoder_20572893348733
// MI455X (gfx1250) — compile-verified, matrix operand order checked by reading, not run on hardware
//
#include <hip/hip_runtime.h>

// MI455X (gfx1250) fused NewsEncoder forward.
// Compute-bound: 4 GEMMs of [64x256]x[256x256] per batch element (275 GFLOP total)
// -> v_wmma_f32_16x16x32_bf16, with weight K-panels delivered by the Tensor Data
// Mover (tensor_load_to_lds, TENSORcnt double buffering). Memory side is trivial
// (51 MB embedding table lives in the 192 MB L2).

typedef __attribute__((ext_vector_type(16))) __bf16          v16bf;
typedef __attribute__((ext_vector_type(8)))  float           v8f;
typedef __attribute__((ext_vector_type(8)))  unsigned short  v8s;
typedef __attribute__((ext_vector_type(16))) unsigned short  v16s;
typedef __attribute__((ext_vector_type(4)))  float           v4f;
typedef __attribute__((ext_vector_type(4)))  unsigned short  v4s;
typedef __attribute__((ext_vector_type(4)))  unsigned int    v4u;
typedef __attribute__((ext_vector_type(8)))  unsigned int    v8u;

#define LSEQ 64
#define DDIM 256
#define NHEAD 8
#define DHEAD 32
#define AS 264          // bf16 row stride (halfs) for 64x256 A-style LDS buffers
#define WS 40           // weight K-panel row stride (halfs): 32 + 4-dword TDM pad
#define HS 260          // f32 row stride for hidden buffer
#define PANEL_HALFS (256 * WS)
#define PANEL_BYTES (PANEL_HALFS * 2)

__device__ __forceinline__ unsigned short f2bf(float f) {
  unsigned int u = __builtin_bit_cast(unsigned int, f);
  u += 0x7FFFu + ((u >> 16) & 1u);           // round-to-nearest-even
  return (unsigned short)(u >> 16);
}
__device__ __forceinline__ float bf2f(unsigned short h) {
  return __builtin_bit_cast(float, ((unsigned int)h) << 16);
}
__device__ __forceinline__ unsigned ldsOff(const void* p) {
  // low 32 bits of a generic pointer into LDS = byte offset in the workgroup's LDS
  return (unsigned)(unsigned long long)p;
}

// TDM: async-load one 256(n) x 32(k) bf16 K-panel (row stride 256 halfs in global,
// column-major weights) into LDS with 16B padding after every 64B row -> WS halfs.
__device__ __forceinline__ void tdm_load_panel(const unsigned short* gsrc,
                                               unsigned lds_byte_off) {
  const unsigned long long ga = (unsigned long long)gsrc;
  v4u g0;
  g0[0] = 1u;                                     // count=1 (valid user descriptor)
  g0[1] = lds_byte_off;                           // lds_addr
  g0[2] = (unsigned)ga;                           // global_addr[31:0]
  g0[3] = ((unsigned)(ga >> 32) & 0x1FFFFFFu)     // global_addr[56:32]
          | (2u << 30);                           // type = 2 ("image")
  v8u g1;
  g1[0] = (1u << 16)        // data_size = 1 -> 2 bytes
        | (1u << 20)        // pad_enable
        | (3u << 22)        // pad_interval = 16 dwords (one 64B tile row)
        | (3u << 25);       // pad_amount   = 4 dwords (16B) -> 80B LDS row stride
  g1[1] = (32u  << 16);     // tensor_dim0 = 32  ([79:48] lo half)
  g1[2] = (256u << 16);     // tensor_dim0 hi = 0 | tensor_dim1 = 256 lo
  g1[3] = (32u  << 16);     // tensor_dim1 hi = 0 | tile_dim0 = 32
  g1[4] = 256u;             // tile_dim1 = 256, tile_dim2 = 0
  g1[5] = 256u;             // tensor_dim0_stride = 256 elements (lo 32)
  g1[6] = 0u;
  g1[7] = 0u;
  asm volatile("tensor_load_to_lds %0, %1" :: "s"(g0), "s"(g1) : "memory");
}

// One 64x256 = A[64x256](LDS bf16) @ Wt(global bf16, column-major: wt[n*256+k]).
// 8 waves: wave w -> M-tile (w&3), N-tiles [(w>>2)*8, +8). K loop: 8 steps of 32,
// weight panels TDM-double-buffered in LDS.
template<int OUT_F32>
__device__ __forceinline__ void gemm_64x256(
    const unsigned short* __restrict__ A,
    const unsigned short* __restrict__ Wt,
    unsigned short* __restrict__ outB,   // LDS bf16 [64][AS]   (if !OUT_F32)
    float*          __restrict__ outF,   // LDS f32  [64][HS]   (if OUT_F32)
    unsigned short* __restrict__ wpan,   // LDS [2][256][WS] K-panels
    int tid)
{
  const int wv = tid >> 5;
  const int ln = tid & 31;
  const int mt = wv & 3;
  const int nh = wv >> 2;
  const int lm = ln & 15;
  const int kh = ln >> 4;

  v8f acc[8];
#pragma unroll
  for (int j = 0; j < 8; ++j) acc[j] = (v8f){0.f,0.f,0.f,0.f,0.f,0.f,0.f,0.f};

  const unsigned wpB = ldsOff(wpan);
  if (wv == 0) tdm_load_panel(Wt, wpB);           // panel 0 -> buf 0

#pragma unroll
  for (int ks = 0; ks < 8; ++ks) {
    if (wv == 0) {
      if (ks < 7) {
        // prefetch next panel into the other buffer (last read 2 barriers ago)
        tdm_load_panel(Wt + (ks + 1) * 32, wpB + ((ks + 1) & 1) * PANEL_BYTES);
        __builtin_amdgcn_s_wait_tensorcnt(1);     // in-order: panel ks has landed
      } else {
        __builtin_amdgcn_s_wait_tensorcnt(0);
      }
    }
    __syncthreads();                              // publish panel ks

    const unsigned short* pan = wpan + (ks & 1) * PANEL_HALFS;

    // A fragment (ISA 16-bit A 16x32 layout): two contiguous 8-half chunks
    const unsigned short* ap = A + (mt * 16 + lm) * AS + ks * 32 + kh * 8;
    v16s av;
    ((v8s*)&av)[0] = *(const v8s*)ap;
    ((v8s*)&av)[1] = *(const v8s*)(ap + 16);
    const v16bf af = __builtin_bit_cast(v16bf, av);

#pragma unroll
    for (int j = 0; j < 8; ++j) {
      const int nt = nh * 8 + j;
      const unsigned short* bp = pan + (nt * 16 + lm) * WS + kh * 8;
      v16s bv;
      ((v8s*)&bv)[0] = *(const v8s*)bp;
      ((v8s*)&bv)[1] = *(const v8s*)(bp + 16);
      const v16bf bfv = __builtin_bit_cast(v16bf, bv);
      acc[j] = __builtin_amdgcn_wmma_f32_16x16x32_bf16(
          false, af, false, bfv, (short)0, acc[j], false, false);
    }
    __syncthreads();                              // reads done -> buffer reusable
  }

  // D tile layout: lane (lm, kh); VGPR v -> row m = mt*16 + kh*8 + v, col n
#pragma unroll
  for (int j = 0; j < 8; ++j) {
    const int n = (nh * 8 + j) * 16 + lm;
#pragma unroll
    for (int v = 0; v < 8; ++v) {
      const int m = mt * 16 + kh * 8 + v;
      if constexpr (OUT_F32) outF[m * HS + n] = acc[j][v];
      else                   outB[m * AS + n] = f2bf(acc[j][v]);
    }
  }
  __syncthreads();
}

// One-shot prep: wt[i][n][k] = bf16(W_i[k][n])  (column-major bf16 weights in d_ws)
__global__ __launch_bounds__(256) void prep_weights(
    const float* __restrict__ WQ, const float* __restrict__ WK,
    const float* __restrict__ WV, const float* __restrict__ WO,
    unsigned short* __restrict__ wt)
{
  const int blk = blockIdx.x;      // 4*256 blocks
  const int i = blk >> 8;
  const int n = blk & 255;
  const int k = threadIdx.x;
  const float* W = (i == 0) ? WQ : (i == 1) ? WK : (i == 2) ? WV : WO;
  wt[i * 65536 + n * 256 + k] = f2bf(W[k * 256 + n]);
}

__global__ __launch_bounds__(256) void newsenc_fused(
    const int*   __restrict__ tok, const float* __restrict__ emb,
    const unsigned short* __restrict__ wt,
    const float* __restrict__ dw,  const float* __restrict__ dbp,
    float*       __restrict__ out)
{
  __shared__ unsigned short xb[LSEQ * AS];        // x, later context (bf16)
  __shared__ unsigned short wpan[2 * PANEL_HALFS];// TDM K-panels; small arrays overlay
  __shared__ unsigned short qkv[3 * LSEQ * AS];   // Q|K|V bf16; later f32 hidden

  const int tid = threadIdx.x;
  const int b   = blockIdx.x;

  // ---- Phase A: embedding gather + f32->bf16 into LDS --------------------
  {
    const int l = tid >> 2;
    const int q = tid & 3;
    const int t = tok[b * LSEQ + l];
    const float* rp = emb + (size_t)t * DDIM + q * 64;
    unsigned short* xp = xb + l * AS + q * 64;
#pragma unroll
    for (int j = 0; j < 16; ++j) {
      v4f f = ((const v4f*)rp)[j];
      v4s h;
      h[0] = f2bf(f[0]); h[1] = f2bf(f[1]); h[2] = f2bf(f[2]); h[3] = f2bf(f[3]);
      *(v4s*)(xp + j * 4) = h;
    }
  }
  __syncthreads();

  unsigned short* Q = qkv;
  unsigned short* K = qkv + LSEQ * AS;
  unsigned short* V = qkv + 2 * LSEQ * AS;
  float* hid = (float*)qkv;        // reuses Q/K/V bytes after they are consumed

  // ---- Phase B: Q/K/V = x @ W* (WMMA bf16, TDM-fed panels) ---------------
  gemm_64x256<0>(xb, wt + 0 * 65536, Q, nullptr, wpan, tid);
  gemm_64x256<0>(xb, wt + 1 * 65536, K, nullptr, wpan, tid);
  gemm_64x256<0>(xb, wt + 2 * 65536, V, nullptr, wpan, tid);

  // ---- Phase C: fastformer attention math (overlays the idle panel LDS) --
  float* gq  = (float*)wpan;       // 256 f32
  float* gk  = gq + 256;           // 256 f32
  float* wsm = gk + 256;           // 64*8 f32
  {  // column means: global_q / global_k
    float sq = 0.f, sk = 0.f;
#pragma unroll 4
    for (int l = 0; l < LSEQ; ++l) {
      sq += bf2f(Q[l * AS + tid]);
      sk += bf2f(K[l * AS + tid]);
    }
    gq[tid] = sq * (1.f / 64.f);
    gk[tid] = sk * (1.f / 64.f);
  }
  __syncthreads();
  for (int p = tid; p < LSEQ * NHEAD; p += 256) {   // logits w[l][h]
    const int l = p >> 3, h = p & 7;
    float s = 0.f;
#pragma unroll
    for (int dh = 0; dh < DHEAD; ++dh) {
      const int dd = h * DHEAD + dh;
      s += gq[dd] * bf2f(K[l * AS + dd]) + gk[dd] * bf2f(Q[l * AS + dd]);
    }
    wsm[l * NHEAD + h] = s;
  }
  __syncthreads();
  if (tid < NHEAD) {                                 // softmax over L per head
    const int h = tid;
    float mx = -1e30f;
    for (int l = 0; l < LSEQ; ++l) mx = fmaxf(mx, wsm[l * NHEAD + h]);
    float ssum = 0.f;
    for (int l = 0; l < LSEQ; ++l) { float e = __expf(wsm[l * NHEAD + h] - mx); wsm[l * NHEAD + h] = e; ssum += e; }
    const float inv = 1.f / ssum;
    for (int l = 0; l < LSEQ; ++l) wsm[l * NHEAD + h] *= inv;
  }
  __syncthreads();
  {  // context = w * V  -> overwrite xb (bf16 A operand for GEMM 4)
    const int l = tid >> 2, q = tid & 3;
#pragma unroll
    for (int j = 0; j < 64; ++j) {
      const int d = q * 64 + j;
      const float cv = wsm[l * NHEAD + (d >> 5)] * bf2f(V[l * AS + d]);
      xb[l * AS + d] = f2bf(cv);
    }
  }
  __syncthreads();

  // ---- Phase D: hidden = context @ WO (f32 out, aliases dead Q/K/V) ------
  gemm_64x256<1>(xb, wt + 3 * 65536, nullptr, hid, wpan, tid);

  // ---- Phase E: scores, softmax over L, weighted sum -> out[b] -----------
  float* spart = (float*)wpan;     // panels are dead now; reuse from base
  float* sc    = spart + 256;
  float* attnv = sc + 64;
  {
    const int l = tid >> 2, q = tid & 3;
    float s = 0.f;
    const float* hp = hid + l * HS + q * 64;
#pragma unroll 8
    for (int j = 0; j < 64; ++j) s += hp[j] * dw[q * 64 + j];
    spart[l * 4 + q] = s;
  }
  __syncthreads();
  if (tid < LSEQ)
    sc[tid] = spart[tid * 4] + spart[tid * 4 + 1] + spart[tid * 4 + 2] + spart[tid * 4 + 3] + dbp[0];
  __syncthreads();
  if (tid == 0) {
    float mx = -1e30f;
    for (int l = 0; l < LSEQ; ++l) mx = fmaxf(mx, sc[l]);
    float ssum = 0.f;
    for (int l = 0; l < LSEQ; ++l) { float e = __expf(sc[l] - mx); attnv[l] = e; ssum += e; }
    const float inv = 1.f / ssum;
    for (int l = 0; l < LSEQ; ++l) attnv[l] *= inv;
  }
  __syncthreads();
  {
    float s = 0.f;
#pragma unroll 8
    for (int l = 0; l < LSEQ; ++l) s += hid[l * HS + tid] * attnv[l];
    out[(size_t)b * DDIM + tid] = s;
  }
}

extern "C" void kernel_launch(void* const* d_in, const int* in_sizes, int n_in,
                              void* d_out, int out_size, void* d_ws, size_t ws_size,
                              hipStream_t stream) {
  (void)in_sizes; (void)n_in; (void)out_size; (void)ws_size;
  const int*   tokens = (const int*)d_in[0];
  const float* emb    = (const float*)d_in[1];
  const float* WQ     = (const float*)d_in[2];
  const float* WK     = (const float*)d_in[3];
  const float* WV     = (const float*)d_in[4];
  const float* WO     = (const float*)d_in[5];
  const float* dw     = (const float*)d_in[6];
  const float* db     = (const float*)d_in[7];
  float* out          = (float*)d_out;
  unsigned short* wt  = (unsigned short*)d_ws;   // 4 * 256*256 bf16 = 512 KB

  prep_weights<<<4 * 256, 256, 0, stream>>>(WQ, WK, WV, WO, wt);
  newsenc_fused<<<8192, 256, 0, stream>>>(tokens, emb, wt, dw, db, out);
}
